// FlaxBloomAttention_86423331930232
// MI455X (gfx1250) — matrix operand order repaired and checked by reading, not compiled, not run on hardware
//
#include <hip/hip_runtime.h>

// ---------------------------------------------------------------------------
// Bloom attention block on MI455X (gfx1250, wave32, WMMA bf16).
// f32->bf16 convert, QKV GEMM (WMMA, reg-prefetch pipeline), flash attention
// (WMMA, block-cooperative LDS K/V staging, split causal tail), output GEMM
// (WMMA) + bias + residual.
// ---------------------------------------------------------------------------

constexpr int B_  = 2;
constexpr int S_  = 2048;
constexpr int H_  = 4096;
constexpr int NH_ = 32;
constexpr int D_  = 128;

typedef __attribute__((ext_vector_type(16))) __bf16 v16bf_t;
typedef __attribute__((ext_vector_type(8)))  float  v8f_t;

union BF16x16 { v16bf_t v; unsigned int u[8]; uint4 q[2]; unsigned short h[16]; };
union F8      { v8f_t   v; float f[8]; };

static __device__ __forceinline__ unsigned short f2bf(float x) {
    unsigned int u = __float_as_uint(x);
    unsigned int r = u + 0x7FFFu + ((u >> 16) & 1u);   // round-to-nearest-even
    return (unsigned short)(r >> 16);
}

// ---------------------------------------------------------------------------
// f32 -> bf16 elementwise convert (2 elements / thread, packed uint store)
// ---------------------------------------------------------------------------
__global__ void cvt_f32_bf16(const float* __restrict__ s,
                             unsigned short* __restrict__ d, long n) {
    long i      = ((long)blockIdx.x * blockDim.x + threadIdx.x) * 2;
    long stride = (long)gridDim.x * blockDim.x * 2;
    for (; i < n; i += stride) {
        float2 f = *reinterpret_cast<const float2*>(s + i);
        unsigned int u = (unsigned int)f2bf(f.x) | ((unsigned int)f2bf(f.y) << 16);
        *reinterpret_cast<unsigned int*>(d + i) = u;
    }
}

// ---------------------------------------------------------------------------
// Tiled bf16 WMMA GEMM: C[M,N] = A[M,K] @ B[K,N] + bias (+ residual)
// Block tile 128x256, 8 waves (2x4), wave tile 64x64 (4x4 WMMA), K-step 32.
// Register-prefetch pipeline: next global tile loads overlap current WMMAs.
// LDS rows padded to 40 shorts (80B pitch) -> 16B aligned, bank-conflict free.
// ---------------------------------------------------------------------------
template <bool F32OUT>
__global__ __launch_bounds__(256) void gemm_bf16(
    const unsigned short* __restrict__ A, const unsigned short* __restrict__ Bm,
    const float* __restrict__ bias, const float* __restrict__ resid,
    float* __restrict__ outF, unsigned short* __restrict__ outH,
    int M, int N, int K) {
    __shared__ unsigned short At[128 * 40];   // [m][k], pitch 40
    __shared__ unsigned short Bt[256 * 40];   // transposed [n][k], pitch 40

    const int t  = threadIdx.x;
    const int w  = t >> 5, L = t & 31;
    const int hi = L >> 4, ln = L & 15;
    const int wm = w >> 2, wn = w & 3;        // 2 waves (M) x 4 waves (N)
    const int m0 = blockIdx.y * 128, n0 = blockIdx.x * 256;

    F8 acc[4][4];
#pragma unroll
    for (int i = 0; i < 4; ++i)
#pragma unroll
        for (int j = 0; j < 4; ++j)
#pragma unroll
            for (int r = 0; r < 8; ++r) acc[i][j].f[r] = 0.0f;

    const int arow = t >> 1, aks = (t & 1) * 16;
    const int bkk  = t >> 3, bns = (t & 7) * 32;
    const unsigned short* aSrcBase = A + (size_t)(m0 + arow) * K + aks;
    const unsigned short* bSrcBase = Bm + (size_t)bkk * N + n0 + bns;

    uint4 apf[2], bpf[4];
    auto prefetch = [&](int k0) {
        const uint4* as = reinterpret_cast<const uint4*>(aSrcBase + k0);
        apf[0] = as[0];
        apf[1] = as[1];
        const uint4* bs = reinterpret_cast<const uint4*>(bSrcBase + (size_t)k0 * N);
#pragma unroll
        for (int q = 0; q < 4; ++q) bpf[q] = bs[q];
    };
    prefetch(0);

    for (int k0 = 0; k0 < K; k0 += 32) {
        // ---- store prefetched tile to LDS ----
        {
            uint4* dst = reinterpret_cast<uint4*>(&At[arow * 40 + aks]);
            dst[0] = apf[0];
            dst[1] = apf[1];
        }
        {
            unsigned short tmp[32];
#pragma unroll
            for (int q = 0; q < 4; ++q)
                *reinterpret_cast<uint4*>(&tmp[q * 8]) = bpf[q];
#pragma unroll
            for (int j = 0; j < 32; ++j) Bt[(bns + j) * 40 + bkk] = tmp[j];
        }
        __syncthreads();

        // ---- issue next tile's global loads (latency hidden by WMMAs) ----
        if (k0 + 32 < K) prefetch(k0 + 32);

        // ---- A fragments (2 x ds_load_b128 each) ----
        BF16x16 af[4];
#pragma unroll
        for (int ms = 0; ms < 4; ++ms) {
            const unsigned short* ar = &At[(wm * 64 + ms * 16 + ln) * 40 + hi * 8];
            af[ms].q[0] = reinterpret_cast<const uint4*>(ar)[0];      // K 0..7 / 8..15
            af[ms].q[1] = *reinterpret_cast<const uint4*>(ar + 16);   // K 16..23 / 24..31
        }
        // ---- B fragments, 4 wmma each ----
#pragma unroll
        for (int ns = 0; ns < 4; ++ns) {
            BF16x16 bfr;
            const unsigned short* br = &Bt[(wn * 64 + ns * 16 + ln) * 40 + hi * 8];
            bfr.q[0] = reinterpret_cast<const uint4*>(br)[0];
            bfr.q[1] = reinterpret_cast<const uint4*>(br)[2];
#pragma unroll
            for (int ms = 0; ms < 4; ++ms)
                acc[ms][ns].v = __builtin_amdgcn_wmma_f32_16x16x32_bf16(
                    false, af[ms].v, false, bfr.v, (short)0, acc[ms][ns].v,
                    false, false);
        }
        __syncthreads();
    }

    // ---- epilogue (compile-time specialized) ----
#pragma unroll
    for (int ms = 0; ms < 4; ++ms)
#pragma unroll
        for (int ns = 0; ns < 4; ++ns) {
            int col = n0 + wn * 64 + ns * 16 + ln;
            float bv = bias[col];
            int row0 = m0 + wm * 64 + ms * 16 + hi * 8;
            size_t base = (size_t)row0 * N + col;
#pragma unroll
            for (int r = 0; r < 8; ++r) {
                size_t idx = base + (size_t)r * N;
                float v = acc[ms][ns].f[r] + bv;
                if (F32OUT) outF[idx] = v + resid[idx];
                else        outH[idx] = f2bf(v);
            }
        }
}

// ---------------------------------------------------------------------------
// Flash attention: block = 8 waves sharing (b, head), q rows [Q0, Q0+128).
// K/V tiles staged cooperatively in LDS; register-prefetch of next tile;
// causal compares only on the single diagonal (tail) tile per wave.
// qkv: bf16 [B, S, NH, 3*D]; ctx: bf16 [B, S, H]
// ---------------------------------------------------------------------------
__global__ __launch_bounds__(256) void attn_kernel(
    const unsigned short* __restrict__ qkv, const float* __restrict__ alibi,
    const int* __restrict__ amask, unsigned short* __restrict__ ctx) {
    const int t  = threadIdx.x;
    const int w  = t >> 5;
    const int L  = t & 31;
    const int hi = L >> 4, ln = L & 15;
    const int b  = blockIdx.z, hh = blockIdx.y;
    const int Q0 = blockIdx.x * 128;
    const int q0 = Q0 + w * 16;

    __shared__ unsigned short Kt[32 * 136];        // [key][chan], pitch 136 (272B)
    __shared__ unsigned short Vt[128 * 40];        // [chan][key], pitch 40  (80B)
    __shared__ unsigned short pshare[8][16][32];   // per-wave P staging

    const size_t tokStride = (size_t)NH_ * 3 * D_;       // 12288
    const size_t headOff   = (size_t)hh * 3 * D_;
    const float  scale     = 0.08838834764831845f;       // 1/sqrt(128)
    const float* alibi_bh  = alibi + ((size_t)b * NH_ + hh) * S_;
    const int*   mask_b    = amask + b * S_;

    // ---- Q fragments: 16x128 bf16 as 4 A-frags, loaded once ----
    BF16x16 aq[4];
    {
        const unsigned short* qrow =
            qkv + ((size_t)(b * S_) + q0 + ln) * tokStride + headOff;
#pragma unroll
        for (int c = 0; c < 4; ++c) {
            const unsigned short* p = qrow + c * 32 + hi * 8;
            aq[c].q[0] = *reinterpret_cast<const uint4*>(p);
            aq[c].q[1] = *reinterpret_cast<const uint4*>(p + 16);
        }
    }

    float m_r[8], l_r[8];
    F8 o[8];
#pragma unroll
    for (int r = 0; r < 8; ++r) { m_r[r] = -3.0e38f; l_r[r] = 0.0f; }
#pragma unroll
    for (int n = 0; n < 8; ++n)
#pragma unroll
        for (int r = 0; r < 8; ++r) o[n].f[r] = 0.0f;

    // staging: 32 keys x 128 chans, 16 elems (32B) per thread per tile
    const int skey = t >> 3, sseg = (t & 7) * 16;
    const unsigned short* kSrcBase =
        qkv + ((size_t)(b * S_) + skey) * tokStride + headOff + D_ + sseg;
    const unsigned short* vSrcBase = kSrcBase + D_;   // V part follows K part

    uint4 kpf[2], vpf[2];
    auto prefetch = [&](int kb) {
        const uint4* ks = reinterpret_cast<const uint4*>(kSrcBase + (size_t)kb * tokStride);
        kpf[0] = ks[0];
        kpf[1] = ks[1];
        const uint4* vs = reinterpret_cast<const uint4*>(vSrcBase + (size_t)kb * tokStride);
        vpf[0] = vs[0];
        vpf[1] = vs[1];
    };
    prefetch(0);

    const int kmax = Q0 + 128;                 // block-uniform loop bound
    for (int kb = 0; kb < kmax; kb += 32) {
        // ---- store prefetched K/V tile to LDS ----
        {
            uint4* dst = reinterpret_cast<uint4*>(&Kt[skey * 136 + sseg]);
            dst[0] = kpf[0];
            dst[1] = kpf[1];
        }
        {
            unsigned short tmp[16];
            *reinterpret_cast<uint4*>(&tmp[0]) = vpf[0];
            *reinterpret_cast<uint4*>(&tmp[8]) = vpf[1];
#pragma unroll
            for (int j = 0; j < 16; ++j) Vt[(sseg + j) * 40 + skey] = tmp[j];
        }
        __syncthreads();

        // ---- issue next tile's global loads ----
        if (kb + 32 < kmax) prefetch(kb + 32);

        if (kb < q0 + 16) {                    // causal guard (wave-uniform)
            // ---- scores: S = Q K^T ----
            F8 sc[2];
#pragma unroll
            for (int g = 0; g < 2; ++g) {
                v8f_t a = {};
#pragma unroll
                for (int c = 0; c < 4; ++c) {
                    BF16x16 bk;
                    const unsigned short* kr =
                        &Kt[(g * 16 + ln) * 136 + c * 32 + hi * 8];
                    bk.q[0] = reinterpret_cast<const uint4*>(kr)[0];
                    bk.q[1] = reinterpret_cast<const uint4*>(kr)[2];
                    a = __builtin_amdgcn_wmma_f32_16x16x32_bf16(
                        false, aq[c].v, false, bk.v, (short)0, a, false, false);
                }
                sc[g].v = a;
            }
            // ---- scale + (alibi + padding-mask) additive bias ----
            const bool tail = (kb + 31 > q0);  // only tile straddling diagonal
#pragma unroll
            for (int g = 0; g < 2; ++g) {
                int key = kb + g * 16 + ln;
                float abm = alibi_bh[key] +
                            ((mask_b[key] != 0) ? 0.0f : -3.0e38f);
#pragma unroll
                for (int r = 0; r < 8; ++r)
                    sc[g].f[r] = fmaf(sc[g].f[r], scale, abm);
                if (tail) {                    // causal select, tail tile only
#pragma unroll
                    for (int r = 0; r < 8; ++r) {
                        int qrow = q0 + r + hi * 8;
                        sc[g].f[r] = (key <= qrow) ? sc[g].f[r] : -3.0e38f;
                    }
                }
            }
            // ---- online softmax (16-lane butterfly per row) ----
            float alpha[8];
#pragma unroll
            for (int r = 0; r < 8; ++r) {
                float mx = fmaxf(sc[0].f[r], sc[1].f[r]);
#pragma unroll
                for (int off = 8; off >= 1; off >>= 1)
                    mx = fmaxf(mx, __shfl_xor(mx, off, 32));
                float mn = fmaxf(m_r[r], mx);
                alpha[r] = __expf(m_r[r] - mn);
                m_r[r] = mn;
                float p0 = __expf(sc[0].f[r] - mn);
                float p1 = __expf(sc[1].f[r] - mn);
                sc[0].f[r] = p0; sc[1].f[r] = p1;
                float sum = p0 + p1;
#pragma unroll
                for (int off = 8; off >= 1; off >>= 1)
                    sum += __shfl_xor(sum, off, 32);
                l_r[r] = l_r[r] * alpha[r] + sum;
            }
#pragma unroll
            for (int n = 0; n < 8; ++n)
#pragma unroll
                for (int r = 0; r < 8; ++r) o[n].f[r] *= alpha[r];

            // ---- re-layout P (C-layout -> A-layout) via per-wave LDS ----
#pragma unroll
            for (int g = 0; g < 2; ++g)
#pragma unroll
                for (int r = 0; r < 8; ++r)
                    pshare[w][r + hi * 8][g * 16 + ln] = f2bf(sc[g].f[r]);
            BF16x16 ap;                        // same-wave DS ops are in-order
            {
                const unsigned short* pr = &pshare[w][ln][hi * 8];
                ap.q[0] = *reinterpret_cast<const uint4*>(pr);
                ap.q[1] = *reinterpret_cast<const uint4*>(pr + 16);
            }
            // ---- O += P * V ----
#pragma unroll
            for (int n = 0; n < 8; ++n) {
                BF16x16 bv;
                const unsigned short* vr = &Vt[(n * 16 + ln) * 40 + hi * 8];
                bv.q[0] = reinterpret_cast<const uint4*>(vr)[0];
                bv.q[1] = reinterpret_cast<const uint4*>(vr)[2];
                o[n].v = __builtin_amdgcn_wmma_f32_16x16x32_bf16(
                    false, ap.v, false, bv.v, (short)0, o[n].v, false, false);
            }
        }
        __syncthreads();
    }

    // ---- finalize: O / l, store bf16 context [B,S,H] ----
#pragma unroll
    for (int r = 0; r < 8; ++r) l_r[r] = 1.0f / l_r[r];
#pragma unroll
    for (int n = 0; n < 8; ++n)
#pragma unroll
        for (int r = 0; r < 8; ++r) {
            int row = q0 + r + hi * 8;
            int d = n * 16 + ln;
            ctx[((size_t)(b * S_) + row) * H_ + hh * D_ + d] =
                f2bf(o[n].f[r] * l_r[r]);
        }
}

// ---------------------------------------------------------------------------
// Host launcher
// ---------------------------------------------------------------------------
extern "C" void kernel_launch(void* const* d_in, const int* in_sizes, int n_in,
                              void* d_out, int out_size, void* d_ws, size_t ws_size,
                              hipStream_t stream) {
    (void)in_sizes; (void)n_in; (void)out_size; (void)ws_size;
    const float* hidden   = (const float*)d_in[0];
    const float* residual = (const float*)d_in[1];
    const float* alibi    = (const float*)d_in[2];
    const int*   amask    = (const int*)d_in[3];
    const float* w_qkv    = (const float*)d_in[4];
    const float* b_qkv    = (const float*)d_in[5];
    const float* w_out    = (const float*)d_in[6];
    const float* b_out    = (const float*)d_in[7];
    float*       out      = (float*)d_out;

    char* ws = (char*)d_ws;
    size_t off = 0;
    auto alloc = [&](size_t bytes) -> void* {
        void* p = ws + off;
        off += (bytes + 255) & ~(size_t)255;
        return p;
    };
    unsigned short* hid_bf  = (unsigned short*)alloc((size_t)B_ * S_ * H_ * 2);
    unsigned short* wqkv_bf = (unsigned short*)alloc((size_t)H_ * 3 * H_ * 2);
    unsigned short* wout_bf = (unsigned short*)alloc((size_t)H_ * H_ * 2);
    unsigned short* qkv_bf  = (unsigned short*)alloc((size_t)B_ * S_ * 3 * H_ * 2);
    unsigned short* ctx_bf  = (unsigned short*)alloc((size_t)B_ * S_ * H_ * 2);

    auto cvt = [&](const float* s, unsigned short* d, long n) {
        int blocks = (int)((n / 2 + 255) / 256);
        cvt_f32_bf16<<<blocks, 256, 0, stream>>>(s, d, n);
    };
    cvt(hidden, hid_bf, (long)B_ * S_ * H_);
    cvt(w_qkv, wqkv_bf, (long)H_ * 3 * H_);
    cvt(w_out, wout_bf, (long)H_ * H_);

    // QKV projection: [4096,4096] @ [4096,12288] + b_qkv -> bf16 [B,S,NH,3D]
    gemm_bf16<false><<<dim3(3 * H_ / 256, (B_ * S_) / 128), 256, 0, stream>>>(
        hid_bf, wqkv_bf, b_qkv, nullptr, nullptr, qkv_bf, B_ * S_, 3 * H_, H_);

    // Flash attention: grid (S/128 q-blocks, heads, batch), 8 waves/block
    attn_kernel<<<dim3(S_ / 128, NH_, B_), 256, 0, stream>>>(
        qkv_bf, alibi, amask, ctx_bf);

    // Output projection + bias + residual -> f32 d_out
    gemm_bf16<true><<<dim3(H_ / 256, (B_ * S_) / 128), 256, 0, stream>>>(
        ctx_bf, wout_bf, b_out, residual, out, nullptr, B_ * S_, H_, H_);
}
